// MinGRU_Layers_16810501996682
// MI455X (gfx1250) — compile-verified, matrix operand order checked by reading, not run on hardware
//
#include <hip/hip_runtime.h>
#include <hip/hip_bf16.h>
#include <math.h>

// Problem constants (from setup_inputs): B=2, S=2048, D=512, V=32000, F=2048
#define BB 2
#define SS 2048
#define DD 512
#define VV 32000
#define FF 2048
#define MTOK (BB * SS)      // 4096 token rows
#define CHUNK 128           // scan chunk length
#define NCH (SS / CHUNK)    // 16 chunks per sequence

typedef __attribute__((ext_vector_type(16))) __bf16 bf16x16;
typedef __attribute__((ext_vector_type(8)))  float  f32x8;

// ---------------------------------------------------------------------------
// Weight prep: f32 (R x C, row-major) -> bf16 transposed (C x R, row-major).
// ---------------------------------------------------------------------------
__global__ __launch_bounds__(256)
void k_transpose_to_bf16(const float* __restrict__ in, __bf16* __restrict__ out,
                         int R, int C) {
  __shared__ __bf16 tile[32][33];
  const int c0 = blockIdx.x * 32, r0 = blockIdx.y * 32;
  const int tx = threadIdx.x, ty = threadIdx.y;
#pragma unroll
  for (int j = 0; j < 32; j += 8)
    tile[ty + j][tx] = (__bf16)in[(size_t)(r0 + ty + j) * C + (c0 + tx)];
  __syncthreads();
#pragma unroll
  for (int j = 0; j < 32; j += 8)
    out[(size_t)(c0 + ty + j) * R + (r0 + tx)] = tile[tx][ty + j];
}

// ---------------------------------------------------------------------------
// Embedding gather + RMSNorm(gamma1) -> f32 residual copy + bf16 GEMM operand.
// ---------------------------------------------------------------------------
__global__ __launch_bounds__(256)
void k_embed_rmsnorm(const int* __restrict__ tok, const float* __restrict__ emb,
                     const float* __restrict__ gamma,
                     float* __restrict__ xn, __bf16* __restrict__ xnb) {
  const int row = blockIdx.x, t = threadIdx.x;
  const float* e = emb + (size_t)tok[row] * DD;
  const float v0 = e[t], v1 = e[t + 256];
  __shared__ float red[256];
  red[t] = v0 * v0 + v1 * v1;
  __syncthreads();
  for (int o = 128; o > 0; o >>= 1) { if (t < o) red[t] += red[t + o]; __syncthreads(); }
  const float n = fmaxf(sqrtf(red[0]), 1e-12f);
  const float s = sqrtf((float)DD) / n;
  const float o0 = v0 * s * (gamma[t] + 1.f);
  const float o1 = v1 * s * (gamma[t + 256] + 1.f);
  xn[(size_t)row * DD + t] = o0;
  xn[(size_t)row * DD + t + 256] = o1;
  xnb[(size_t)row * DD + t] = (__bf16)o0;
  xnb[(size_t)row * DD + t + 256] = (__bf16)o1;
}

__global__ __launch_bounds__(256)
void k_rmsnorm_bf16(const float* __restrict__ in, const float* __restrict__ gamma,
                    __bf16* __restrict__ out) {
  const int row = blockIdx.x, t = threadIdx.x;
  const float* p = in + (size_t)row * DD;
  const float v0 = p[t], v1 = p[t + 256];
  __shared__ float red[256];
  red[t] = v0 * v0 + v1 * v1;
  __syncthreads();
  for (int o = 128; o > 0; o >>= 1) { if (t < o) red[t] += red[t + o]; __syncthreads(); }
  const float n = fmaxf(sqrtf(red[0]), 1e-12f);
  const float s = sqrtf((float)DD) / n;
  out[(size_t)row * DD + t]       = (__bf16)(v0 * s * (gamma[t] + 1.f));
  out[(size_t)row * DD + t + 256] = (__bf16)(v1 * s * (gamma[t + 256] + 1.f));
}

// ---------------------------------------------------------------------------
// bf16 WMMA GEMM with async-to-LDS staged B tile.
//   C[M,N] = A[M,K] * BT[N,K]^T
// Block = 256 threads (8 waves). Block tile 128x128; wave -> 16 rows x 128
// cols (8 accumulators). K stepped 32. The 128x32 bf16 B tile is staged to
// LDS with GLOBAL_LOAD_ASYNC_TO_LDS_B128 (2 per thread, ASYNCcnt tracked),
// double-buffered; consumed by all 8 waves via ds_load_b128 (row stride 56
// elems = 112 B: 16B-aligned, conflict-free across 16 lanes). A fragments
// load straight from global. Last K-step peeled so the loop is branch-free.
// EPI: 0 = f32 store, 1 = bf16(gelu(acc+bias)), 2 = f32(acc+bias+resid)
// ---------------------------------------------------------------------------
#define BROW 56   // padded LDS row stride in bf16 elems (112 bytes)

__device__ __forceinline__ void stage_async_b128(const __bf16* g, __bf16* l) {
  unsigned lds_off =
      (unsigned)(unsigned long long)(__attribute__((address_space(3))) __bf16*)l;
  unsigned long long ga = (unsigned long long)(uintptr_t)g;
  asm volatile("global_load_async_to_lds_b128 %0, %1, off"
               :: "v"(lds_off), "v"(ga) : "memory");
}

template <int EPI>
__global__ __launch_bounds__(256)
void k_gemm_bf16(const __bf16* __restrict__ A, const __bf16* __restrict__ BT,
                 void* __restrict__ Cout, const float* __restrict__ bias,
                 const float* __restrict__ resid, int M, int N, int K) {
  __shared__ __bf16 lbs[2][128 * BROW];

  const int lane = threadIdx.x & 31;
  const int wave = threadIdx.x >> 5;
  const int rowBase = blockIdx.y * 128 + wave * 16;
  const int colBase = blockIdx.x * 128;
  const int half = lane >> 4;
  const int l16  = lane & 15;

  // staging: thread -> rows n = tid/2, k-span [ (tid&1)*16, +16 ) = 2x b128
  const int sn  = threadIdx.x >> 1;
  const int sk0 = (threadIdx.x & 1) * 16;
  const __bf16* bsrc = BT + (size_t)(colBase + sn) * K + sk0;
  __bf16* ldst0 = &lbs[0][sn * BROW + sk0];
  __bf16* ldst1 = &lbs[1][sn * BROW + sk0];

  f32x8 acc[8] = {};

  // A layout (ISA 7.12.2, 16-bit A 16x32): lanes 0-15 hold K {0..7,16..23},
  // lanes 16-31 hold K {8..15,24..31} for row M = lane%16.
  const __bf16* aRow = A + (size_t)(rowBase + l16) * K + half * 8;

  // prologue: stage first B tile into buffer 0
  stage_async_b128(bsrc, ldst0);
  stage_async_b128(bsrc + 8, ldst0 + 8);

  const int kLast = K - 32;
  for (int k0 = 0; k0 < kLast; k0 += 32) {
    const int buf = (k0 >> 5) & 1;
    __bf16* nl = buf ? ldst0 : ldst1;
    stage_async_b128(bsrc + k0 + 32, nl);
    stage_async_b128(bsrc + k0 + 40, nl + 8);
    asm volatile("s_wait_asynccnt 0x2" ::: "memory");  // tile k0 landed
    __syncthreads();                                    // visible to all waves

    union { bf16x16 f; uint4 u[2]; } ua;
    ua.u[0] = *(const uint4*)(aRow + k0);
    ua.u[1] = *(const uint4*)(aRow + k0 + 16);
    const bf16x16 afrag = ua.f;

#pragma unroll
    for (int j = 0; j < 8; ++j) {
      const __bf16* lp = &lbs[buf][(j * 16 + l16) * BROW + half * 8];
      union { bf16x16 f; uint4 u[2]; } ub;
      ub.u[0] = *(const uint4*)lp;          // ds_load_b128
      ub.u[1] = *(const uint4*)(lp + 16);   // ds_load_b128
      acc[j] = __builtin_amdgcn_wmma_f32_16x16x32_bf16(
          false, afrag, false, ub.f, (short)0, acc[j], false, false);
    }
    __syncthreads();  // reads done before this buffer is restaged
  }

  // peeled last K-step
  {
    const int buf = (kLast >> 5) & 1;
    asm volatile("s_wait_asynccnt 0x0" ::: "memory");
    __syncthreads();
    union { bf16x16 f; uint4 u[2]; } ua;
    ua.u[0] = *(const uint4*)(aRow + kLast);
    ua.u[1] = *(const uint4*)(aRow + kLast + 16);
    const bf16x16 afrag = ua.f;
#pragma unroll
    for (int j = 0; j < 8; ++j) {
      const __bf16* lp = &lbs[buf][(j * 16 + l16) * BROW + half * 8];
      union { bf16x16 f; uint4 u[2]; } ub;
      ub.u[0] = *(const uint4*)lp;
      ub.u[1] = *(const uint4*)(lp + 16);
      acc[j] = __builtin_amdgcn_wmma_f32_16x16x32_bf16(
          false, afrag, false, ub.f, (short)0, acc[j], false, false);
    }
  }

  // C/D layout: VGPR r, lanes 0-15 -> (M=r, N=lane); lanes 16-31 -> (M=8+r).
#pragma unroll
  for (int j = 0; j < 8; ++j) {
    const int col = colBase + j * 16 + l16;
#pragma unroll
    for (int r = 0; r < 8; ++r) {
      const int row = rowBase + half * 8 + r;
      const size_t o = (size_t)row * N + col;
      const float v = acc[j][r];
      if (EPI == 0) {
        ((float*)Cout)[o] = v;
      } else if (EPI == 1) {
        const float x = v + bias[col];
        const float g = 0.5f * x * (1.f + erff(x * 0.70710678118654752f));
        ((__bf16*)Cout)[o] = (__bf16)g;
      } else {
        ((float*)Cout)[o] = v + bias[col] + resid[o];
      }
    }
  }
}

// ---------------------------------------------------------------------------
// minGRU scan, chunked 3-pass parallel form of the Heinsen scan.
// h_t = c_t*h_{t-1} + v_t, c = sigmoid(-gate), v = exp(log_z + log_g), h_0=0.
// ---------------------------------------------------------------------------
__device__ __forceinline__ float softplus_f(float x) {
  return (x > 15.f) ? x : log1pf(expf(x));
}

__global__ __launch_bounds__(256)
void k_scan_local(const float* __restrict__ hg, float* __restrict__ hloc,
                  float* __restrict__ Acum, float* __restrict__ Pc,
                  float* __restrict__ Hc) {
  const int bx = blockIdx.x;           // BB * NCH * 2 blocks
  const int b  = bx >> 5;
  const int r  = bx & 31;
  const int ch = r >> 1;
  const int d  = ((r & 1) << 8) + threadIdx.x;
  float h = 0.f, A = 1.f;
  for (int t = 0; t < CHUNK; ++t) {
    const size_t row = (size_t)b * SS + ch * CHUNK + t;
    const float hid = hg[row * (2 * DD) + d];
    const float gat = hg[row * (2 * DD) + DD + d];
    const float sp    = softplus_f(gat);
    const float log_z = gat - sp;                 // -softplus(-gate)
    const float logg  = (hid >= 0.f) ? logf(hid + 0.5f) : -softplus_f(-hid);
    const float v = expf(log_z + logg);
    const float c = expf(-sp);                    // 1 - z
    h = fmaf(c, h, v);
    A *= c;
    hloc[row * DD + d] = h;
    Acum[row * DD + d] = A;
  }
  const size_t si = ((size_t)b * NCH + ch) * DD + d;
  Pc[si] = A;
  Hc[si] = h;
}

__global__ __launch_bounds__(512)
void k_scan_combine(const float* __restrict__ Pc, const float* __restrict__ Hc,
                    float* __restrict__ start, float* __restrict__ next_hidden) {
  const int b = blockIdx.x, d = threadIdx.x;
  float s = 0.f;
#pragma unroll
  for (int ch = 0; ch < NCH; ++ch) {
    const size_t si = ((size_t)b * NCH + ch) * DD + d;
    start[si] = s;
    s = fmaf(Pc[si], s, Hc[si]);
  }
  next_hidden[(size_t)b * DD + d] = s;   // gru output at t = S-1
}

__global__ __launch_bounds__(256)
void k_scan_fix(const float* __restrict__ hloc, const float* __restrict__ Acum,
                const float* __restrict__ start, const float* __restrict__ xn,
                float* __restrict__ x, __bf16* __restrict__ xb) {
  const int bx = blockIdx.x;
  const int b  = bx >> 5;
  const int r  = bx & 31;
  const int ch = r >> 1;
  const int d  = ((r & 1) << 8) + threadIdx.x;
  const float s = start[((size_t)b * NCH + ch) * DD + d];
  for (int t = 0; t < CHUNK; ++t) {
    const size_t row = (size_t)b * SS + ch * CHUNK + t;
    const float h = fmaf(Acum[row * DD + d], s, hloc[row * DD + d]);
    const float o = h + xn[row * DD + d];
    x[row * DD + d]  = o;
    xb[row * DD + d] = (__bf16)o;
  }
}

// ---------------------------------------------------------------------------
// Loss: per-row logsumexp, then deterministic single-block mean.
// ---------------------------------------------------------------------------
__global__ __launch_bounds__(256)
void k_row_loss(const float* __restrict__ logits, const int* __restrict__ labels,
                float* __restrict__ row_loss) {
  const int row = blockIdx.x, t = threadIdx.x;
  const float* l = logits + (size_t)row * VV;
  __shared__ float red[256];
  float m = -INFINITY;
  for (int i = t; i < VV; i += 256) m = fmaxf(m, l[i]);
  red[t] = m; __syncthreads();
  for (int o = 128; o > 0; o >>= 1) { if (t < o) red[t] = fmaxf(red[t], red[t + o]); __syncthreads(); }
  m = red[0]; __syncthreads();
  float s = 0.f;
  for (int i = t; i < VV; i += 256) s += expf(l[i] - m);
  red[t] = s; __syncthreads();
  for (int o = 128; o > 0; o >>= 1) { if (t < o) red[t] += red[t + o]; __syncthreads(); }
  if (t == 0) row_loss[row] = (logf(red[0]) + m) - l[labels[row]];
}

__global__ __launch_bounds__(256)
void k_final_loss(const float* __restrict__ row_loss, float* __restrict__ loss) {
  const int t = threadIdx.x;
  float s = 0.f;
  for (int i = t; i < MTOK; i += 256) s += row_loss[i];
  __shared__ float red[256];
  red[t] = s; __syncthreads();
  for (int o = 128; o > 0; o >>= 1) { if (t < o) red[t] += red[t + o]; __syncthreads(); }
  if (t == 0) loss[0] = red[0] / (float)MTOK;
}

// ---------------------------------------------------------------------------
extern "C" void kernel_launch(void* const* d_in, const int* in_sizes, int n_in,
                              void* d_out, int out_size, void* d_ws, size_t ws_size,
                              hipStream_t stream) {
  (void)in_sizes; (void)n_in; (void)out_size; (void)ws_size;
  const int*   inputs   = (const int*)  d_in[0];
  const int*   labels   = (const int*)  d_in[1];
  const float* emb      = (const float*)d_in[2];
  const float* w_gru    = (const float*)d_in[3];
  const float* gamma1   = (const float*)d_in[4];
  const float* gamma2   = (const float*)d_in[5];
  const float* w1       = (const float*)d_in[6];
  const float* b1       = (const float*)d_in[7];
  const float* w2       = (const float*)d_in[8];
  const float* b2       = (const float*)d_in[9];
  const float* w_logits = (const float*)d_in[10];

  float* out_loss = (float*)d_out;
  float* logits   = out_loss + 1;                         // (4096, 32000) f32
  float* next_hid = logits + (size_t)MTOK * VV;           // (B, 1, D) f32

  char* ws = (char*)d_ws;
  size_t off = 0;
  auto alloc = [&](size_t bytes) -> void* {
    off = (off + 255) & ~(size_t)255;
    void* p = ws + off;
    off += bytes;
    return p;
  };
  float*  xn    = (float*) alloc((size_t)MTOK * DD * 4);
  __bf16* xnb   = (__bf16*)alloc((size_t)MTOK * DD * 2);
  float*  hg    = (float*) alloc((size_t)MTOK * 2 * DD * 4);
  float*  xf    = (float*) alloc((size_t)MTOK * DD * 4);   // pass1 hloc, pass3 x (in-place)
  float*  Acum  = (float*) alloc((size_t)MTOK * DD * 4);
  __bf16* xfb   = (__bf16*)alloc((size_t)MTOK * DD * 2);
  __bf16* h1b   = (__bf16*)alloc((size_t)MTOK * FF * 2);
  float*  x2    = (float*) alloc((size_t)MTOK * DD * 4);
  __bf16* yb    = (__bf16*)alloc((size_t)MTOK * DD * 2);
  __bf16* wgruT = (__bf16*)alloc((size_t)2 * DD * DD * 2);
  __bf16* w1T   = (__bf16*)alloc((size_t)FF * DD * 2);
  __bf16* w2T   = (__bf16*)alloc((size_t)DD * FF * 2);
  __bf16* wlogT = (__bf16*)alloc((size_t)VV * DD * 2);
  float*  Pc    = (float*) alloc((size_t)BB * NCH * DD * 4);
  float*  Hc    = (float*) alloc((size_t)BB * NCH * DD * 4);
  float*  stc   = (float*) alloc((size_t)BB * NCH * DD * 4);
  float*  rloss = (float*) alloc((size_t)MTOK * 4);

  const dim3 tb(32, 8);
  k_transpose_to_bf16<<<dim3((2 * DD) / 32, DD / 32), tb, 0, stream>>>(w_gru, wgruT, DD, 2 * DD);
  k_transpose_to_bf16<<<dim3(FF / 32, DD / 32), tb, 0, stream>>>(w1, w1T, DD, FF);
  k_transpose_to_bf16<<<dim3(DD / 32, FF / 32), tb, 0, stream>>>(w2, w2T, FF, DD);
  k_transpose_to_bf16<<<dim3(VV / 32, DD / 32), tb, 0, stream>>>(w_logits, wlogT, DD, VV);

  // 1) x = rms_norm(emb[inputs], gamma1)
  k_embed_rmsnorm<<<MTOK, 256, 0, stream>>>(inputs, emb, gamma1, xn, xnb);

  // 2) hg = x @ w_gru   (4096 x 1024, K=512)
  k_gemm_bf16<0><<<dim3((2 * DD) / 128, MTOK / 128), 256, 0, stream>>>(
      xnb, wgruT, hg, nullptr, nullptr, MTOK, 2 * DD, DD);

  // 3) minGRU chunked scan + residual; emits next_hidden
  k_scan_local<<<BB * NCH * 2, 256, 0, stream>>>(hg, xf, Acum, Pc, Hc);
  k_scan_combine<<<BB, 512, 0, stream>>>(Pc, Hc, stc, next_hid);
  k_scan_fix<<<BB * NCH * 2, 256, 0, stream>>>(xf, Acum, stc, xn, xf, xfb);

  // 4) h1 = gelu(x @ w1 + b1)   (4096 x 2048, K=512), bf16 out
  k_gemm_bf16<1><<<dim3(FF / 128, MTOK / 128), 256, 0, stream>>>(
      xfb, w1T, h1b, b1, nullptr, MTOK, FF, DD);

  // 5) x2 = h1 @ w2 + b2 + x    (4096 x 512, K=2048), f32 out
  k_gemm_bf16<2><<<dim3(DD / 128, MTOK / 128), 256, 0, stream>>>(
      h1b, w2T, x2, b2, xf, MTOK, DD, FF);

  // 6) y = rms_norm(x2, gamma2)
  k_rmsnorm_bf16<<<MTOK, 256, 0, stream>>>(x2, gamma2, yb);

  // 7) logits = y @ w_logits    (4096 x 32000, K=512)
  k_gemm_bf16<0><<<dim3(VV / 128, MTOK / 128), 256, 0, stream>>>(
      yb, wlogT, logits, nullptr, nullptr, MTOK, VV, DD);

  // 8) loss = mean(lse - logit[label])
  k_row_loss<<<MTOK, 256, 0, stream>>>(logits, labels, rloss);
  k_final_loss<<<1, 256, 0, stream>>>(rloss, out_loss);
}
